// AttentiveFP_42417097015328
// MI455X (gfx1250) — matrix operand order repaired and checked by reading, not compiled, not run on hardware
//
#include <hip/hip_runtime.h>
#include <hip/hip_bf16.h>

typedef __attribute__((ext_vector_type(16))) _Float16 v16h;
typedef __attribute__((ext_vector_type(8)))  _Float16 v8h;
typedef __attribute__((ext_vector_type(8)))  float    v8f;
typedef __attribute__((ext_vector_type(4)))  unsigned int v4ui;
typedef __attribute__((ext_vector_type(8)))  int      v8si;
typedef __attribute__((ext_vector_type(4)))  int      v4si;

#define BATCH 256
#define LA    128
#define MNB   6
#define AFD   39
#define BFD   10
#define DD    128
#define RR    3
#define TT    2
#define UU    64
#define NEGV  (-9e8f)

#if defined(__gfx1250__) && __has_builtin(__builtin_amdgcn_tensor_load_to_lds)
#define HAVE_TDM 1
#else
#define HAVE_TDM 0
#endif

__device__ __forceinline__ float lreluf(float v){ return v > 0.f ? v : 0.01f * v; }
__device__ __forceinline__ float eluf (float v){ return v > 0.f ? v : (expf(v) - 1.f); }
__device__ __forceinline__ float sigmf(float v){ return 1.f / (1.f + expf(-v)); }

// ---------------------------------------------------------------------------
// WMMA GEMM: out = act(A_f16[rows,K] @ B_f16[K,N] + bias)
//   K%32==0, N%64==0, rows%32==0. Row-major A and B.
//   Whole B panel (K*N f16, <=96KB) is TDM-DMA'd into LDS once per workgroup;
//   each wave computes a 32x64 tile (2 row tiles x 4 col tiles, 8 wmma accs).
//   A fragment (16x32): lane m = row0+(lane&15); two contiguous 8xf16 runs.
//   B fragment (32x16): lane holds B col (lane&15), k order as in A; 16 f16 strided by N.
// ---------------------------------------------------------------------------
template<int ACT>
__global__ void __launch_bounds__(256)
gemm_wmma_f16(const _Float16* __restrict__ A, const _Float16* __restrict__ Bm,
              const float* __restrict__ bias, float* __restrict__ out32,
              _Float16* __restrict__ out16, int rows, int K, int N)
{
    extern __shared__ _Float16 bsh[];
    const int lane = threadIdx.x & 31;
    const int wave = threadIdx.x >> 5;

#if HAVE_TDM
    // Stage full B panel into LDS with the Tensor Data Mover (wave 0 issues).
    if (wave == 0) {
        unsigned long long ga = (unsigned long long)Bm;
        unsigned ldsb = (unsigned)(unsigned long long)&bsh[0];
        v4ui g0 = {};
        g0[0] = 1u;                                    // count=1 (valid descriptor)
        g0[1] = ldsb;                                  // lds_addr
        g0[2] = (unsigned)ga;                          // global_addr[31:0]
        g0[3] = (unsigned)((ga >> 32) & 0x01FFFFFFu)   // global_addr[56:32]
              | (2u << 30);                            // type=2 ("image")
        v8si g1 = {};
        g1[0] = (int)(1u << 16);                       // data_size=1 -> 2 bytes
        g1[1] = (int)(((unsigned)N & 0xFFFFu) << 16);  // tensor_dim0[15:0]
        g1[2] = (int)(((unsigned)N >> 16) | (((unsigned)K & 0xFFFFu) << 16)); // dim0 hi | dim1 lo
        g1[3] = (int)(((unsigned)K >> 16) | (((unsigned)N & 0xFFFFu) << 16)); // dim1 hi | tile_dim0=N
        g1[4] = (int)((unsigned)K & 0xFFFFu);          // tile_dim1=K, tile_dim2=0
        g1[5] = (int)N;                                // tensor_dim0_stride = N
        g1[6] = 0; g1[7] = 0;
        v4si z4 = {};
#if __clang_major__ >= 23
        v8si z8 = {};
        __builtin_amdgcn_tensor_load_to_lds(g0, g1, z4, z4, z8, 0);
#else
        __builtin_amdgcn_tensor_load_to_lds(g0, g1, z4, z4, 0);
#endif
        __builtin_amdgcn_s_wait_tensorcnt(0);
    }
    __syncthreads();
#endif

    const int nc = N >> 6;                              // 64-wide column tiles
    const long totalWaves = (long)(rows >> 5) * nc;     // 32-row wave tiles
    const long gw = (long)blockIdx.x * 8 + wave;
    if (gw >= totalWaves) return;
    const int row0 = (int)((gw / nc) << 5);
    const int col0 = (int)((gw % nc) << 6);

    const int kh = (lane >> 4) << 3;                    // 0 or 8
    const _Float16* aRow0 = A + (size_t)(row0 + (lane & 15)) * K + kh;
    const _Float16* aRow1 = aRow0 + (size_t)16 * K;
#if HAVE_TDM
    const _Float16* bRow = bsh + (size_t)kh * N + col0 + (lane & 15);   // DS-pipe reads
#else
    const _Float16* bRow = Bm + (size_t)kh * N + col0 + (lane & 15);    // fallback: L2 reads
#endif

    v8f acc[8] = {};
    for (int k0 = 0; k0 < K; k0 += 32) {
        v16h a0, a1;
        *reinterpret_cast<v8h*>(&a0)       = *reinterpret_cast<const v8h*>(aRow0 + k0);
        *(reinterpret_cast<v8h*>(&a0) + 1) = *reinterpret_cast<const v8h*>(aRow0 + k0 + 16);
        *reinterpret_cast<v8h*>(&a1)       = *reinterpret_cast<const v8h*>(aRow1 + k0);
        *(reinterpret_cast<v8h*>(&a1) + 1) = *reinterpret_cast<const v8h*>(aRow1 + k0 + 16);
        if (k0 + 32 < K) {
            __builtin_prefetch((const void*)(aRow0 + k0 + 32), 0, 0);
            __builtin_prefetch((const void*)(aRow1 + k0 + 32), 0, 0);
        }
        const _Float16* bp = bRow + (size_t)k0 * N;
        #pragma unroll
        for (int j = 0; j < 4; ++j) {
            v16h b;
            #pragma unroll
            for (int s = 0; s < 8; ++s) {
                b[s]     = bp[(size_t)s * N + (j << 4)];
                b[8 + s] = bp[(size_t)(16 + s) * N + (j << 4)];
            }
            acc[j]     = __builtin_amdgcn_wmma_f32_16x16x32_f16(false, a0, false, b, (short)0, acc[j],     false, false);
            acc[4 + j] = __builtin_amdgcn_wmma_f32_16x16x32_f16(false, a1, false, b, (short)0, acc[4 + j], false, false);
        }
    }

    // C/D layout: VGPR r, lane l -> row = r + (l>=16?8:0), col = l&15
    const int nbase = col0 + (lane & 15);
    const int mb0   = row0 + ((lane >> 4) << 3);
    #pragma unroll
    for (int rt = 0; rt < 2; ++rt) {
        #pragma unroll
        for (int j = 0; j < 4; ++j) {
            const int n  = nbase + (j << 4);
            const float bv = bias ? bias[n] : 0.f;
            v8f av = acc[rt * 4 + j];
            #pragma unroll
            for (int r = 0; r < 8; ++r) {
                float v = av[r] + bv;
                if (ACT == 1) v = lreluf(v);
                size_t o = (size_t)(mb0 + rt * 16 + r) * N + n;
                if (out32) out32[o] = v;
                if (out16) out16[o] = (_Float16)v;
            }
        }
    }
}

// ---------------------------------------------------------------------------
// Conversion / packing kernels
// ---------------------------------------------------------------------------
__global__ void f32_to_f16_k(const float* __restrict__ in, _Float16* __restrict__ out, int n)
{
    int i = blockIdx.x * 256 + threadIdx.x;
    if (i < n) out[i] = (_Float16)in[i];
}

__global__ void pad_weight_k(const float* __restrict__ in, _Float16* __restrict__ out,
                             int Kin, int Kp, int N)
{
    int i = blockIdx.x * 256 + threadIdx.x;
    if (i >= Kp * N) return;
    int k = i / N, n = i - k * N;
    out[i] = (k < Kin) ? (_Float16)in[k * N + n] : (_Float16)0.f;
}

__global__ void pad_x_atom_k(const float* __restrict__ xa, _Float16* __restrict__ out)
{
    int i = blockIdx.x * 256 + threadIdx.x;
    if (i >= BATCH * LA * 64) return;
    int c = i & 63, bl = i >> 6;
    out[i] = (c < AFD) ? (_Float16)xa[(size_t)bl * AFD + c] : (_Float16)0.f;
}

__global__ void build_nbin_k(const float* __restrict__ xa, const float* __restrict__ xb,
                             const int* __restrict__ ai, const int* __restrict__ bi,
                             _Float16* __restrict__ out)
{
    size_t i = (size_t)blockIdx.x * 256 + threadIdx.x;
    if (i >= (size_t)BATCH * LA * MNB * 64) return;
    int c = (int)(i & 63);
    size_t blm = i >> 6;
    int b = (int)(blm / (LA * MNB));
    _Float16 v = (_Float16)0.f;
    if (c < AFD) {
        int idx = ai[blm];
        v = (_Float16)xa[((size_t)b * LA + idx) * AFD + c];
    } else if (c < AFD + BFD) {
        int idx = bi[blm];
        v = (_Float16)xb[((size_t)b * 2 * LA + idx) * BFD + (c - AFD)];
    }
    out[i] = v;
}

__global__ void gather_nbr_k(const _Float16* __restrict__ cur16, const int* __restrict__ ai,
                             _Float16* __restrict__ nbr16)
{
    size_t i = (size_t)blockIdx.x * 256 + threadIdx.x;
    if (i >= (size_t)BATCH * LA * MNB * DD) return;
    int c = (int)(i & 127);
    size_t blm = i >> 7;
    size_t bl  = blm / MNB;
    int b = (int)(bl >> 7);
    int idx = ai[blm];
    nbr16[i] = cur16[((size_t)b * LA + idx) * DD + c];
}

// ---------------------------------------------------------------------------
// Atom-level attention: score(M=6 nbrs) -> softmax -> weighted attend -> elu
// ---------------------------------------------------------------------------
__global__ void __launch_bounds__(128)
attn_ctx_k(const float* __restrict__ cur32, const _Float16* __restrict__ nbr16,
           const float* __restrict__ attendv, const int* __restrict__ aidx,
           const float* __restrict__ alW, const float* __restrict__ alB,
           float* __restrict__ ctx32, _Float16* __restrict__ ctx16)
{
    __shared__ float red[128];
    __shared__ float aw[MNB];
    const int bl = blockIdx.x;
    const int c  = threadIdx.x;

    float curv = cur32[(size_t)bl * DD + c];
    red[c] = curv * alW[c]; __syncthreads();
    for (int s = 64; s; s >>= 1) { if (c < s) red[c] += red[c + s]; __syncthreads(); }
    float curdot = red[0]; __syncthreads();

    for (int mm = 0; mm < MNB; ++mm) {
        float nv = (float)nbr16[((size_t)bl * MNB + mm) * DD + c];
        red[c] = nv * alW[DD + c]; __syncthreads();
        for (int s = 64; s; s >>= 1) { if (c < s) red[c] += red[c + s]; __syncthreads(); }
        if (c == 0) {
            int idx = aidx[bl * MNB + mm];
            aw[mm] = lreluf(curdot + red[0] + alB[0]) + ((idx == LA - 1) ? NEGV : 0.f);
        }
        __syncthreads();
    }
    if (c == 0) {
        float mx = aw[0];
        for (int mm = 1; mm < MNB; ++mm) mx = fmaxf(mx, aw[mm]);
        float e[MNB], s = 0.f;
        for (int mm = 0; mm < MNB; ++mm) { e[mm] = expf(aw[mm] - mx); s += e[mm]; }
        float inv = 1.f / s;
        for (int mm = 0; mm < MNB; ++mm) {
            int idx = aidx[bl * MNB + mm];
            aw[mm] = e[mm] * inv * ((idx == LA - 1) ? 0.f : 1.f);
        }
    }
    __syncthreads();
    float ctx = 0.f;
    for (int mm = 0; mm < MNB; ++mm)
        ctx += aw[mm] * attendv[((size_t)bl * MNB + mm) * DD + c];
    ctx = eluf(ctx);
    ctx32[(size_t)bl * DD + c] = ctx;
    ctx16[(size_t)bl * DD + c] = (_Float16)ctx;
}

// ---------------------------------------------------------------------------
// GRU pointwise: h' = (1-z)*n + z*h ; act = relu(h')
// ---------------------------------------------------------------------------
__global__ void __launch_bounds__(256)
gru_elem_k(const float* __restrict__ gi, const float* __restrict__ gh,
           float* __restrict__ h32, _Float16* __restrict__ h16,
           float* __restrict__ act32, _Float16* __restrict__ act16, int rows)
{
    size_t i = (size_t)blockIdx.x * 256 + threadIdx.x;
    if (i >= (size_t)rows * DD) return;
    size_t row = i >> 7;
    int c = (int)(i & 127);
    size_t g = row * (3 * DD) + c;
    float r  = sigmf(gi[g]        + gh[g]);
    float z  = sigmf(gi[g + DD]   + gh[g + DD]);
    float n  = tanhf(gi[g + 2*DD] + r * gh[g + 2*DD]);
    float hn = (1.f - z) * n + z * h32[i];
    h32[i] = hn;
    h16[i] = (_Float16)hn;
    float a = hn > 0.f ? hn : 0.f;
    act32[i] = a;
    if (act16) act16[i] = (_Float16)a;
}

// ---------------------------------------------------------------------------
// mol_feature init: masked sum over atoms
// ---------------------------------------------------------------------------
__global__ void __launch_bounds__(128)
mol_init_k(const float* __restrict__ cur32, const float* __restrict__ mask,
           float* __restrict__ molf32, _Float16* __restrict__ molf16,
           float* __restrict__ act32)
{
    int b = blockIdx.x, c = threadIdx.x;
    float s = 0.f;
    for (int l = 0; l < LA; ++l)
        s += cur32[((size_t)b * LA + l) * DD + c] * mask[b * LA + l];
    molf32[b * DD + c] = s;
    molf16[b * DD + c] = (_Float16)s;
    act32 [b * DD + c] = s > 0.f ? s : 0.f;
}

// ---------------------------------------------------------------------------
// mol-level attention: softmax over L=128 atoms. One block per molecule.
// ---------------------------------------------------------------------------
__global__ void __launch_bounds__(128)
mol_attn_k(const float* __restrict__ actmol, const float* __restrict__ cur32,
           const float* __restrict__ attendm, const float* __restrict__ mask,
           const float* __restrict__ mW, const float* __restrict__ mB,
           float* __restrict__ mctx)
{
    __shared__ float red[128];
    __shared__ float aw[128];
    const int b = blockIdx.x, t = threadIdx.x;

    red[t] = actmol[(size_t)b * DD + t] * mW[t]; __syncthreads();
    for (int s = 64; s; s >>= 1) { if (t < s) red[t] += red[t + s]; __syncthreads(); }
    float amdot = red[0]; __syncthreads();

    float d2 = 0.f;
    const float* curRow = cur32 + ((size_t)b * LA + t) * DD;
    for (int c = 0; c < DD; ++c) d2 += curRow[c] * mW[DD + c];
    float mk = mask[b * LA + t];
    float sc = lreluf(amdot + d2 + mB[0]) + ((mk == 0.f) ? NEGV : 0.f);

    red[t] = sc; __syncthreads();
    for (int s = 64; s; s >>= 1) { if (t < s) red[t] = fmaxf(red[t], red[t + s]); __syncthreads(); }
    float mx = red[0]; __syncthreads();
    float e = expf(sc - mx);
    red[t] = e; __syncthreads();
    for (int s = 64; s; s >>= 1) { if (t < s) red[t] += red[t + s]; __syncthreads(); }
    float inv = 1.f / red[0]; __syncthreads();
    aw[t] = e * inv * mk; __syncthreads();

    float ctx = 0.f;
    for (int l = 0; l < LA; ++l)
        ctx += aw[l] * attendm[((size_t)b * LA + l) * DD + t];
    mctx[(size_t)b * DD + t] = eluf(ctx);
}

// ---------------------------------------------------------------------------
// BatchNorm (train-mode stats over B=256), per channel
// ---------------------------------------------------------------------------
__global__ void __launch_bounds__(128)
bn_k(const float* __restrict__ x, const float* __restrict__ gamma,
     const float* __restrict__ beta, float* __restrict__ y32,
     _Float16* __restrict__ y16)
{
    int c = threadIdx.x;
    float s = 0.f, ss = 0.f;
    for (int b = 0; b < BATCH; ++b) {
        float v = x[(size_t)b * DD + c];
        s += v; ss += v * v;
    }
    float mu  = s / BATCH;
    float var = ss / BATCH - mu * mu;
    float g = gamma[c] * rsqrtf(var + 1e-5f);
    for (int b = 0; b < BATCH; ++b) {
        float v = g * (x[(size_t)b * DD + c] - mu) + beta[c];
        y32[(size_t)b * DD + c] = v;
        y16[(size_t)b * DD + c] = (_Float16)v;
    }
}

// ---------------------------------------------------------------------------
// Output head: xo = (molf @ Wmo + bmo) @ Wo + bo ; out = [xo ; sigmoid(xo)]
// ---------------------------------------------------------------------------
__global__ void __launch_bounds__(256)
out_k(const float* __restrict__ molf, const float* __restrict__ Wmo,
      const float* __restrict__ bmo, const float* __restrict__ Wo,
      const float* __restrict__ bo, float* __restrict__ out)
{
    int b = threadIdx.x;
    float xo = bo[0];
    for (int j = 0; j < UU; ++j) {
        float s = bmo[j];
        for (int c = 0; c < DD; ++c) s += molf[(size_t)b * DD + c] * Wmo[c * UU + j];
        xo += s * Wo[j];
    }
    out[b] = xo;
    out[BATCH + b] = sigmf(xo);
}

// ---------------------------------------------------------------------------
extern "C" void kernel_launch(void* const* d_in, const int* in_sizes, int n_in,
                              void* d_out, int out_size, void* d_ws, size_t ws_size,
                              hipStream_t stream)
{
    (void)in_sizes; (void)n_in; (void)out_size; (void)ws_size;
    const float* x_atom = (const float*)d_in[0];
    const float* x_bond = (const float*)d_in[1];
    const int*   aidx   = (const int*)  d_in[2];
    const int*   bidx   = (const int*)  d_in[3];
    const float* xmask  = (const float*)d_in[4];
    const float* atomW  = (const float*)d_in[6];
    const float* atomB  = (const float*)d_in[7];
    const float* nbrW   = (const float*)d_in[8];
    const float* nbrB   = (const float*)d_in[9];
    const float* gWih   = (const float*)d_in[10];
    const float* gWhh   = (const float*)d_in[11];
    const float* gBih   = (const float*)d_in[12];
    const float* gBhh   = (const float*)d_in[13];
    const float* alW    = (const float*)d_in[14];
    const float* alB    = (const float*)d_in[15];
    const float* atW    = (const float*)d_in[16];
    const float* atB    = (const float*)d_in[17];
    const float* mWih   = (const float*)d_in[18];
    const float* mWhh   = (const float*)d_in[19];
    const float* mBih   = (const float*)d_in[20];
    const float* mBhh   = (const float*)d_in[21];
    const float* malW   = (const float*)d_in[22];
    const float* malB   = (const float*)d_in[23];
    const float* matW   = (const float*)d_in[24];
    const float* matB   = (const float*)d_in[25];
    const float* bng    = (const float*)d_in[26];
    const float* bnb    = (const float*)d_in[27];
    const float* moW    = (const float*)d_in[28];
    const float* moB    = (const float*)d_in[29];
    const float* oW     = (const float*)d_in[30];
    const float* oB     = (const float*)d_in[31];

    const int BL  = BATCH * LA;     // 32768
    const int BLM = BL * MNB;       // 196608

    char* base = (char*)d_ws;
    size_t off = 0;
    auto alloc = [&](size_t bytes) -> void* {
        void* p = base + off;
        off += (bytes + 255) & ~(size_t)255;
        return p;
    };

    _Float16* Wa16   = (_Float16*)alloc((size_t)64 * DD * 2);
    _Float16* Wn16   = (_Float16*)alloc((size_t)64 * DD * 2);
    _Float16* attW16 = (_Float16*)alloc((size_t)RR * DD * DD * 2);
    _Float16* Wih16  = (_Float16*)alloc((size_t)RR * DD * 3 * DD * 2);
    _Float16* Whh16  = (_Float16*)alloc((size_t)RR * DD * 3 * DD * 2);
    _Float16* matW16 = (_Float16*)alloc((size_t)DD * DD * 2);
    _Float16* mWih16 = (_Float16*)alloc((size_t)DD * 3 * DD * 2);
    _Float16* mWhh16 = (_Float16*)alloc((size_t)DD * 3 * DD * 2);
    _Float16* xa16   = (_Float16*)alloc((size_t)BL * 64 * 2);
    _Float16* nbin16 = (_Float16*)alloc((size_t)BLM * 64 * 2);
    _Float16* nbr16  = (_Float16*)alloc((size_t)BLM * DD * 2);
    float*    big    = (float*)   alloc((size_t)BLM * DD * 4);   // attendv | gi+gh | attendm/gim/ghm
    float*    h32    = (float*)   alloc((size_t)BL * DD * 4);
    _Float16* h16    = (_Float16*)alloc((size_t)BL * DD * 2);
    float*    cur32  = (float*)   alloc((size_t)BL * DD * 4);
    _Float16* cur16  = (_Float16*)alloc((size_t)BL * DD * 2);
    float*    ctx32  = (float*)   alloc((size_t)BL * DD * 4);
    _Float16* ctx16  = (_Float16*)alloc((size_t)BL * DD * 2);
    float*    molf32 = (float*)   alloc((size_t)BATCH * DD * 4);
    _Float16* molf16 = (_Float16*)alloc((size_t)BATCH * DD * 2);
    float*    am32   = (float*)   alloc((size_t)BATCH * DD * 4);
    float*    mctx   = (float*)   alloc((size_t)BATCH * DD * 4);
    float*    mbn32  = (float*)   alloc((size_t)BATCH * DD * 4);
    _Float16* mbn16  = (_Float16*)alloc((size_t)BATCH * DD * 2);

    float* attendv = big;
    float* gi      = big;
    float* gh      = big + (size_t)BL * 3 * DD;
    float* attendm = big;
    float* gim     = big;
    float* ghm     = big + (size_t)BATCH * 3 * DD;

    const int TPB = 256;
    auto blocks1d = [](size_t n) { return (unsigned)((n + 255) / 256); };
    auto gemmBlocks = [](long rows, int N) {
        long waves = (rows >> 5) * (N >> 6);            // 32x64 wave tiles
        return (unsigned)((waves + 7) / 8);
    };
    auto smemB = [](int K, int N) { return (size_t)K * N * 2; };  // B panel in LDS

    // --- weight conversion / packing ---
    pad_weight_k<<<blocks1d(64 * DD), TPB, 0, stream>>>(atomW, Wa16, AFD, 64, DD);
    pad_weight_k<<<blocks1d(64 * DD), TPB, 0, stream>>>(nbrW,  Wn16, AFD + BFD, 64, DD);
    f32_to_f16_k<<<blocks1d(RR * DD * DD),     TPB, 0, stream>>>(atW,  attW16, RR * DD * DD);
    f32_to_f16_k<<<blocks1d(RR * DD * 3 * DD), TPB, 0, stream>>>(gWih, Wih16,  RR * DD * 3 * DD);
    f32_to_f16_k<<<blocks1d(RR * DD * 3 * DD), TPB, 0, stream>>>(gWhh, Whh16,  RR * DD * 3 * DD);
    f32_to_f16_k<<<blocks1d(DD * DD),          TPB, 0, stream>>>(matW, matW16, DD * DD);
    f32_to_f16_k<<<blocks1d(DD * 3 * DD),      TPB, 0, stream>>>(mWih, mWih16, DD * 3 * DD);
    f32_to_f16_k<<<blocks1d(DD * 3 * DD),      TPB, 0, stream>>>(mWhh, mWhh16, DD * 3 * DD);

    // --- input packing (pad K to 64 for WMMA) ---
    pad_x_atom_k<<<blocks1d((size_t)BL * 64), TPB, 0, stream>>>(x_atom, xa16);
    build_nbin_k<<<blocks1d((size_t)BLM * 64), TPB, 0, stream>>>(x_atom, x_bond, aidx, bidx, nbin16);

    // --- atom_fc: h = cur = lrelu(x_atom @ Wa + b) ---
    gemm_wmma_f16<1><<<gemmBlocks(BL, DD), 256, smemB(64, DD), stream>>>(
        xa16, Wa16, atomB, h32, h16, BL, 64, DD);
    hipMemcpyAsync(cur32, h32, (size_t)BL * DD * 4, hipMemcpyDeviceToDevice, stream);
    hipMemcpyAsync(cur16, h16, (size_t)BL * DD * 2, hipMemcpyDeviceToDevice, stream);

    // --- neighbor_fc: nbr = lrelu(nbin @ Wn + b), kept f16 ---
    gemm_wmma_f16<1><<<gemmBlocks(BLM, DD), 256, smemB(64, DD), stream>>>(
        nbin16, Wn16, nbrB, nullptr, nbr16, BLM, 64, DD);

    // --- radius loop ---
    for (int d = 0; d < RR; ++d) {
        gemm_wmma_f16<0><<<gemmBlocks(BLM, DD), 256, smemB(DD, DD), stream>>>(
            nbr16, attW16 + (size_t)d * DD * DD, atB + d * DD, attendv, nullptr, BLM, DD, DD);
        attn_ctx_k<<<BL, 128, 0, stream>>>(cur32, nbr16, attendv, aidx,
                                           alW + d * 2 * DD, alB + d, ctx32, ctx16);
        gemm_wmma_f16<0><<<gemmBlocks(BL, 3 * DD), 256, smemB(DD, 3 * DD), stream>>>(
            ctx16, Wih16 + (size_t)d * DD * 3 * DD, gBih + d * 3 * DD, gi, nullptr, BL, DD, 3 * DD);
        gemm_wmma_f16<0><<<gemmBlocks(BL, 3 * DD), 256, smemB(DD, 3 * DD), stream>>>(
            h16, Whh16 + (size_t)d * DD * 3 * DD, gBhh + d * 3 * DD, gh, nullptr, BL, DD, 3 * DD);
        gru_elem_k<<<blocks1d((size_t)BL * DD), 256, 0, stream>>>(gi, gh, h32, h16, cur32, cur16, BL);
        if (d + 1 < RR)
            gather_nbr_k<<<blocks1d((size_t)BLM * DD), 256, 0, stream>>>(cur16, aidx, nbr16);
    }

    // --- molecule stage ---
    mol_init_k<<<BATCH, 128, 0, stream>>>(cur32, xmask, molf32, molf16, am32);
    for (int t = 0; t < TT; ++t) {
        gemm_wmma_f16<0><<<gemmBlocks(BL, DD), 256, smemB(DD, DD), stream>>>(
            cur16, matW16, matB, attendm, nullptr, BL, DD, DD);
        mol_attn_k<<<BATCH, 128, 0, stream>>>(am32, cur32, attendm, xmask, malW, malB, mctx);
        bn_k<<<1, 128, 0, stream>>>(mctx, bng, bnb, mbn32, mbn16);
        gemm_wmma_f16<0><<<gemmBlocks(BATCH, 3 * DD), 256, smemB(DD, 3 * DD), stream>>>(
            mbn16, mWih16, mBih, gim, nullptr, BATCH, DD, 3 * DD);
        gemm_wmma_f16<0><<<gemmBlocks(BATCH, 3 * DD), 256, smemB(DD, 3 * DD), stream>>>(
            molf16, mWhh16, mBhh, ghm, nullptr, BATCH, DD, 3 * DD);
        gru_elem_k<<<blocks1d((size_t)BATCH * DD), 256, 0, stream>>>(
            gim, ghm, molf32, molf16, am32, nullptr, BATCH);
    }

    out_k<<<1, 256, 0, stream>>>(molf32, moW, moB, oW, oB, (float*)d_out);
}
